// Model_67705864454755
// MI455X (gfx1250) — matrix operand order repaired and checked by reading, not compiled, not run on hardware
//
#include <hip/hip_runtime.h>
#include <hip/hip_bf16.h>
#include <math.h>

#define MODEL_DIM 64
#define N_ATOMS   512
#define KEEP      64
#define FB_K      256
#define NBLOCKS   13
#define TLEN      32768
#define BATCH     4
#define TN        64      // time columns per GEMM workgroup
#define CHUNK_T   64      // time columns per top-k stage-1 chunk

typedef __attribute__((ext_vector_type(16))) __bf16 v16bf;
typedef __attribute__((ext_vector_type(8)))  float  v8f;
typedef __attribute__((ext_vector_type(4)))  unsigned int v4u;
typedef __attribute__((ext_vector_type(8)))  int    v8i;
typedef __attribute__((ext_vector_type(4)))  int    v4i;

union FragU { unsigned int u[8]; v16bf v; };

__device__ __forceinline__ unsigned short f2bf(float f) {
  unsigned int u = __float_as_uint(f);
  unsigned int r = u + 0x7FFFu + ((u >> 16) & 1u);   // round-to-nearest-even
  return (unsigned short)(r >> 16);
}

// Issue a TDM copy of nDwords DWORDs from global (contiguous) into LDS at byte
// offset ldsOff (dynamic-LDS base == 0 since no static __shared__ is used).
// D# per CDNA5 ISA: count=1, data_size=4B, 1-row tile.
__device__ __forceinline__ void tdm_load(const void* gptr, unsigned int ldsOff,
                                         unsigned int nDwords) {
  unsigned long long ga = (unsigned long long)(size_t)gptr;
  v4u g0;
  g0.x = 1u;                                         // count=1, no gather
  g0.y = ldsOff;                                     // lds_addr
  g0.z = (unsigned int)ga;                           // global_addr[31:0]
  g0.w = ((unsigned int)(ga >> 32) & 0x1FFFFFFu) | 0x80000000u; // addr[56:32] | type=2
  v8i g1;
  g1[0] = 0x00020000;                                // wg_mask=0, data_size=4B
  g1[1] = (int)(nDwords << 16);                      // tensor_dim0[15:0]
  g1[2] = (int)((nDwords >> 16) | (1u << 16));       // tensor_dim0[31:16], tensor_dim1=1
  g1[3] = (int)(nDwords << 16);                      // tile_dim0 = nDwords
  g1[4] = 1;                                         // tile_dim1=1, tile_dim2=0
  g1[5] = (int)nDwords;                              // tensor_dim0_stride[31:0]
  g1[6] = 0;
  g1[7] = 0;
  v4i gz = {0, 0, 0, 0};
#if defined(__clang_major__) && __clang_major__ >= 23
  v8i gz8 = {0, 0, 0, 0, 0, 0, 0, 0};
  __builtin_amdgcn_tensor_load_to_lds(g0, g1, gz, gz, gz8, 0);
#else
  __builtin_amdgcn_tensor_load_to_lds(g0, g1, gz, gz, 0);
#endif
}

// A-matrix 16x32 bf16 fragment: lanes 0-15 -> K[kk0..kk0+15],
// lanes 16-31 -> K[kk0+16..+31]; two bf16 per VGPR, loaded as b32.
__device__ __forceinline__ v16bf ldsLoadA(const unsigned short* A, int stride,
                                          int m0, int kk0, int lane) {
  FragU f;
  int m  = m0 + (lane & 15);
  int hi = lane >> 4;
  const unsigned short* row = A + m * stride;
  int kb  = kk0 + hi * 16;
  int kb2 = kk0 + 8 + hi * 16;
#pragma unroll
  for (int v = 0; v < 4; ++v) f.u[v]     = *(const unsigned int*)(row + kb  + 2 * v);
#pragma unroll
  for (int v = 0; v < 4; ++v) f.u[4 + v] = *(const unsigned int*)(row + kb2 + 2 * v);
  return f.v;
}

// B-matrix 32x16 bf16 fragment: lane&15 = N column; lanes 0-15 K[kk0..+15],
// lanes 16-31 K[kk0+16..+31]; LDS holds B as [n][k] so K pairs are contiguous.
__device__ __forceinline__ v16bf ldsLoadB(const unsigned short* B, int stride,
                                          int n0, int kk0, int lane) {
  FragU f;
  int n  = n0 + (lane & 15);
  int kb = kk0 + (lane >> 4) * 16;
  const unsigned short* row = B + n * stride;
#pragma unroll
  for (int v = 0; v < 8; ++v) f.u[v] = *(const unsigned int*)(row + kb + 2 * v);
  return f.v;
}

// ------------- one-time weight pack: f32 -> bf16 GEMM panels in workspace -------------
// wA1: [13][128][192] (scale||gate, kk=k*64+c)  wA2: [13][128][64] (out||next)
// wFb: [64][256]
__global__ void wprep_kernel(const float* __restrict__ sw, const float* __restrict__ gw,
                             const float* __restrict__ ow, const float* __restrict__ nw,
                             const float* __restrict__ fbw,
                             unsigned short* __restrict__ wA1,
                             unsigned short* __restrict__ wA2,
                             unsigned short* __restrict__ wFb) {
  const int nA1 = NBLOCKS * 128 * 192;
  const int nA2 = NBLOCKS * 128 * 64;
  const int nFb = 64 * 256;
  int i = blockIdx.x * blockDim.x + threadIdx.x;
  if (i < nA1) {
    int blk = i / (128 * 192), r = i % (128 * 192);
    int m = r / 192, kk = r % 192;
    int k = kk >> 6, c = kk & 63;
    float wv = (m < 64) ? sw[(((size_t)blk * 64 + m) * 64 + c) * 3 + k]
                        : gw[(((size_t)blk * 64 + (m - 64)) * 64 + c) * 3 + k];
    wA1[i] = f2bf(wv);
  } else if (i < nA1 + nA2) {
    int j = i - nA1;
    int blk = j / (128 * 64), r = j % (128 * 64);
    int m = r >> 6, c = r & 63;
    float wv = (m < 64) ? ow[((size_t)blk * 64 + m) * 64 + c]
                        : nw[((size_t)blk * 64 + (m - 64)) * 64 + c];
    wA2[j] = f2bf(wv);
  } else if (i < nA1 + nA2 + nFb) {
    int j = i - nA1 - nA2;
    wFb[j] = f2bf(fbw[j]);
  }
}

// ---------------- filterbank: x[B,1,T] (k=256, pad 127/128) -> act[B,T,64] -------------
__global__ void fb_kernel(const float* __restrict__ x,
                          const unsigned short* __restrict__ wFb,
                          float* __restrict__ act) {
  extern __shared__ char smem[];
  unsigned short* Aw = (unsigned short*)smem;     // [64][256] bf16 weights (TDM-filled)
  unsigned short* Bx = Aw + 64 * 256;             // [64][256] bf16 im2col
  int wg = blockIdx.x;
  int b  = wg / (TLEN / TN);
  int t0 = (wg % (TLEN / TN)) * TN;
  int tid = threadIdx.x;

  if (tid < 32) {                                 // wave 0: async weight panel fill
    tdm_load(wFb, 0u, (64 * 256 * 2) / 4);
    __builtin_amdgcn_s_wait_tensorcnt(0);
  }
  const float* xb = x + (size_t)b * TLEN;
  for (int i = tid; i < TN * 256; i += 256) {
    int n = i >> 8, k = i & 255;
    int tp = t0 + n + k - 127;
    float v = (tp >= 0 && tp < TLEN) ? xb[tp] : 0.0f;
    Bx[n * 256 + k] = f2bf(v);
  }
  __syncthreads();

  int wave = tid >> 5, lane = tid & 31, hi = lane >> 4;
  int mtile = wave & 3;           // 4 M-tiles cover 64 output channels
  int nt0   = (wave >> 2) * 2;    // 2 waves split the 4 N-tiles
#pragma unroll
  for (int nn = 0; nn < 2; ++nn) {
    int nt = nt0 + nn;
    v8f acc = {};
#pragma unroll
    for (int ks = 0; ks < 8; ++ks) {
      v16bf a  = ldsLoadA(Aw, 256, mtile * 16, ks * 32, lane);
      v16bf bb = ldsLoadB(Bx, 256, nt * 16,    ks * 32, lane);
      acc = __builtin_amdgcn_wmma_f32_16x16x32_bf16(false, a, false, bb,
                                                    (short)0, acc, false, false);
    }
    int t = t0 + nt * 16 + (lane & 15);
#pragma unroll
    for (int v = 0; v < 8; ++v) {
      int m = mtile * 16 + v + 8 * hi;
      act[((size_t)b * TLEN + t) * 64 + m] = acc[v];
    }
  }
}

// ---------------- one WaveNet block: fused scale/gate conv GEMM + out/next GEMM --------
__global__ void block_kernel(const float* __restrict__ src, float* __restrict__ dst,
                             float* __restrict__ outAcc,
                             const unsigned short* __restrict__ wA1,
                             const unsigned short* __restrict__ wA2,
                             const float* __restrict__ sb, const float* __restrict__ gb,
                             const float* __restrict__ ob, const float* __restrict__ nb,
                             int dil) {
  extern __shared__ char smem[];
  unsigned short* A1 = (unsigned short*)smem;   // [128][192] conv weights (TDM-filled)
  unsigned short* B1 = A1 + 128 * 192;          // [64][192]  im2col (kk = k*64+c)
  unsigned short* A2 = B1 + 64 * 192;           // [128][64]  out||next weights (TDM)
  unsigned short* B2 = A2 + 128 * 64;           // [64][64]   h (bf16)
  float*          pre = (float*)(B2 + 64 * 64); // [128][64]  GEMM1 result (f32)

  int wg = blockIdx.x;
  int b  = wg / (TLEN / TN);
  int t0 = (wg % (TLEN / TN)) * TN;
  int tid = threadIdx.x;

  if (tid < 32) {                               // wave 0: async weight panel fills
    tdm_load(wA1, 0u, (128 * 192 * 2) / 4);
    tdm_load(wA2, (unsigned int)(128 * 192 * 2), (128 * 64 * 2) / 4);
    __builtin_amdgcn_s_wait_tensorcnt(0);
  }

  const float* sB = src + (size_t)b * TLEN * 64;
  for (int i = tid; i < 64 * 192; i += 256) {   // im2col (taps t-d, t, t+d)
    int n = i / 192, kk = i % 192;
    int k = kk >> 6, c = kk & 63;
    int tp = t0 + n + (k - 1) * dil;
    float v = (tp >= 0 && tp < TLEN) ? sB[(size_t)tp * 64 + c] : 0.0f;
    B1[i] = f2bf(v);
  }
  __syncthreads();

  int wave = tid >> 5, lane = tid & 31, hi = lane >> 4;
  int m0 = wave * 16;                                     // 8 waves cover M=128
  {
    v8f acc[4] = {{}, {}, {}, {}};
#pragma unroll
    for (int ks = 0; ks < 6; ++ks) {                      // K = 192
      v16bf a = ldsLoadA(A1, 192, m0, ks * 32, lane);
#pragma unroll
      for (int nt = 0; nt < 4; ++nt) {
        v16bf bb = ldsLoadB(B1, 192, nt * 16, ks * 32, lane);
        acc[nt] = __builtin_amdgcn_wmma_f32_16x16x32_bf16(false, a, false, bb,
                                                          (short)0, acc[nt], false, false);
      }
    }
#pragma unroll
    for (int nt = 0; nt < 4; ++nt) {
      int n = nt * 16 + (lane & 15);
#pragma unroll
      for (int v = 0; v < 8; ++v) pre[(m0 + v + 8 * hi) * 64 + n] = acc[nt][v];
    }
  }
  __syncthreads();

  for (int i = tid; i < 64 * 64; i += 256) {              // h = tanh(s)*sigmoid(g)
    int n = i >> 6, c = i & 63;
    float s = tanhf(pre[c * 64 + n] + sb[c]);
    float gv = pre[(c + 64) * 64 + n] + gb[c];
    float g = 1.0f / (1.0f + __expf(-gv));
    B2[n * 64 + c] = f2bf(s * g);
  }
  __syncthreads();

  {
    v8f acc[4] = {{}, {}, {}, {}};
#pragma unroll
    for (int ks = 0; ks < 2; ++ks) {                      // K = 64
      v16bf a = ldsLoadA(A2, 64, m0, ks * 32, lane);
#pragma unroll
      for (int nt = 0; nt < 4; ++nt) {
        v16bf bb = ldsLoadB(B2, 64, nt * 16, ks * 32, lane);
        acc[nt] = __builtin_amdgcn_wmma_f32_16x16x32_bf16(false, a, false, bb,
                                                          (short)0, acc[nt], false, false);
      }
    }
    float* dB = dst    + (size_t)b * TLEN * 64;
    float* oB = outAcc + (size_t)b * TLEN * 64;
#pragma unroll
    for (int nt = 0; nt < 4; ++nt) {
      int t = t0 + nt * 16 + (lane & 15);
#pragma unroll
      for (int v = 0; v < 8; ++v) {
        int m = m0 + v + 8 * hi;
        if (m < 64) {                                     // uniform per wave
          oB[(size_t)t * 64 + m] += acc[nt][v] + ob[m];
        } else {
          int c = m - 64;                                 // next = gemm + nb + residual
          dB[(size_t)t * 64 + c] = acc[nt][v] + nb[c] + sB[(size_t)t * 64 + c];
        }
      }
    }
  }
}

// -------- top-k stage 1: e for a (512 atoms x 64 times) chunk, keep chunk top-64 -------
__global__ void upsel_kernel(const float* __restrict__ outAcc,
                             const float* __restrict__ upw, const float* __restrict__ upb,
                             float* __restrict__ candV, unsigned int* __restrict__ candI) {
  extern __shared__ char smem[];
  float* outs = (float*)smem;                 // [64][64] staged outputs tile
  float* ev   = outs + 64 * 64;               // [512*64] e values
  float* rV   = ev + N_ATOMS * CHUNK_T;       // [256] reduce scratch
  unsigned int* rI = (unsigned int*)(rV + 256);
  int b = blockIdx.x / (TLEN / CHUNK_T);
  int chunk = blockIdx.x % (TLEN / CHUNK_T);
  int t0 = chunk * CHUNK_T;
  int tid = threadIdx.x;

  const float* oB = outAcc + ((size_t)b * TLEN + t0) * 64;
  for (int i = tid; i < 64 * 64; i += 256) outs[i] = oB[i];
  __syncthreads();

  for (int i = tid; i < N_ATOMS * CHUNK_T; i += 256) {
    int a = i >> 6, n = i & 63;
    float acc = upb[a];
    const float* wr = upw + a * 64;
    const float* orow = outs + n * 64;
#pragma unroll
    for (int c = 0; c < 64; ++c) acc += wr[c] * orow[c];
    ev[i] = acc;
  }
  __syncthreads();

  for (int r = 0; r < KEEP; ++r) {
    float best = -INFINITY; unsigned int bi = 0;
    for (int i = tid; i < N_ATOMS * CHUNK_T; i += 256) {
      float v = ev[i];
      if (v > best) { best = v; bi = (unsigned int)i; }
    }
    rV[tid] = best; rI[tid] = bi;
    __syncthreads();
    for (int s = 128; s > 0; s >>= 1) {
      if (tid < s && (rV[tid + s] > rV[tid] ||
                      (rV[tid + s] == rV[tid] && rI[tid + s] < rI[tid]))) {
        rV[tid] = rV[tid + s]; rI[tid] = rI[tid + s];
      }
      __syncthreads();
    }
    if (tid == 0) {
      unsigned int i = rI[0];
      int a = (int)(i >> 6), n = (int)(i & 63);
      size_t o = ((size_t)b * (TLEN / CHUNK_T) + chunk) * KEEP + r;
      candV[o] = rV[0];
      candI[o] = ((unsigned int)a << 16) | (unsigned int)(t0 + n);
      ev[i] = -INFINITY;
    }
    __syncthreads();
  }
}

// -------- top-k stage 2: final top-64 per batch from 32768 candidates ------------------
__global__ void __launch_bounds__(1024)
topk_final(const float* __restrict__ candV, const unsigned int* __restrict__ candI,
           float* __restrict__ topV, unsigned int* __restrict__ topI) {
  extern __shared__ char smem[];
  const int NC = (TLEN / CHUNK_T) * KEEP;     // 32768
  float* v = (float*)smem;
  unsigned int* ix = (unsigned int*)(v + NC);
  float* rV = (float*)(ix + NC);
  unsigned int* rI = (unsigned int*)(rV + 1024);
  int b = blockIdx.x, tid = threadIdx.x;
  for (int i = tid; i < NC; i += 1024) {
    v[i]  = candV[(size_t)b * NC + i];
    ix[i] = candI[(size_t)b * NC + i];
  }
  __syncthreads();
  for (int r = 0; r < KEEP; ++r) {
    float best = -INFINITY; unsigned int bi = 0;
    for (int i = tid; i < NC; i += 1024) {
      if (v[i] > best) { best = v[i]; bi = (unsigned int)i; }
    }
    rV[tid] = best; rI[tid] = bi;
    __syncthreads();
    for (int s = 512; s > 0; s >>= 1) {
      if (tid < s && (rV[tid + s] > rV[tid] ||
                      (rV[tid + s] == rV[tid] && rI[tid + s] < rI[tid]))) {
        rV[tid] = rV[tid + s]; rI[tid] = rI[tid + s];
      }
      __syncthreads();
    }
    if (tid == 0) {
      topV[b * KEEP + r] = rV[0];
      topI[b * KEEP + r] = ix[rI[0]];
      v[rI[0]] = -INFINITY;
    }
    __syncthreads();
  }
}

// -------- sparse down-projection: dst = down_b + sum over selected (a,t) rank-1 --------
__global__ void down_kernel(const float* __restrict__ topV, const unsigned int* __restrict__ topI,
                            const float* __restrict__ dw, const float* __restrict__ db,
                            float* __restrict__ dst) {
  int wg = blockIdx.x;
  int b  = wg / (TLEN / 64);
  int t0 = (wg % (TLEN / 64)) * 64;
  int tid = threadIdx.x;
  for (int i = tid; i < 64 * 64; i += 256) {
    int n = i >> 6, c = i & 63;
    int t = t0 + n;
    float acc = db[c];
    for (int r = 0; r < KEEP; ++r) {
      unsigned int p = topI[b * KEEP + r];
      if ((int)(p & 0xFFFFu) == t) {
        int a = (int)(p >> 16);
        acc += dw[c * N_ATOMS + a] * topV[b * KEEP + r];
      }
    }
    dst[((size_t)b * TLEN + t) * 64 + c] = acc;
  }
}

__global__ void zero_kernel(float* __restrict__ p, size_t n) {
  size_t i = (size_t)blockIdx.x * blockDim.x + threadIdx.x;
  if (i < n) p[i] = 0.0f;
}

__global__ void final_kernel(const float* __restrict__ outAcc, float* __restrict__ out,
                             size_t n) {
  size_t i = (size_t)blockIdx.x * blockDim.x + threadIdx.x;
  if (i >= n) return;
  const float* r = outAcc + i * 64;
  float s = 0.0f;
#pragma unroll
  for (int c = 0; c < 64; ++c) s += r[c];
  out[i] = s;
}

extern "C" void kernel_launch(void* const* d_in, const int* in_sizes, int n_in,
                              void* d_out, int out_size, void* d_ws, size_t ws_size,
                              hipStream_t stream) {
  const float* x   = (const float*)d_in[0];
  const float* fbw = (const float*)d_in[1];
  const float* sw  = (const float*)d_in[2];
  const float* sb  = (const float*)d_in[3];
  const float* gw  = (const float*)d_in[4];
  const float* gb  = (const float*)d_in[5];
  const float* ow  = (const float*)d_in[6];
  const float* ob  = (const float*)d_in[7];
  const float* nw  = (const float*)d_in[8];
  const float* nb  = (const float*)d_in[9];
  const float* upw = (const float*)d_in[10];
  const float* upb = (const float*)d_in[11];
  const float* dw  = (const float*)d_in[12];
  const float* db  = (const float*)d_in[13];
  float* out = (float*)d_out;

  char* ws = (char*)d_ws;
  const size_t actBytes = (size_t)BATCH * TLEN * 64 * sizeof(float);   // 33.5 MB each
  float*        actA   = (float*)(ws);
  float*        actB   = (float*)(ws + actBytes);
  float*        outAcc = (float*)(ws + 2 * actBytes);
  const size_t  candN  = (size_t)BATCH * (TLEN / CHUNK_T) * KEEP;      // 131072
  size_t off = 3 * actBytes;
  float*        candV  = (float*)(ws + off); off += candN * 4;
  unsigned int* candI  = (unsigned int*)(ws + off); off += candN * 4;
  float*        topV   = (float*)(ws + off); off += BATCH * KEEP * 4;
  unsigned int* topI   = (unsigned int*)(ws + off); off += BATCH * KEEP * 4;
  off = (off + 255) & ~(size_t)255;
  unsigned short* wA1  = (unsigned short*)(ws + off); off += (size_t)NBLOCKS * 128 * 192 * 2;
  unsigned short* wA2  = (unsigned short*)(ws + off); off += (size_t)NBLOCKS * 128 * 64 * 2;
  unsigned short* wFb  = (unsigned short*)(ws + off); off += (size_t)64 * 256 * 2;

  static const int DILS_[NBLOCKS] = {1, 3, 9, 27, 81, 243, 1, 243, 81, 27, 9, 3, 1};
  const dim3 blk(256);
  const int gT = BATCH * (TLEN / TN);                     // 2048 workgroups
  const size_t nTot = (size_t)BATCH * TLEN * 64;

  const size_t fbLds  = (size_t)(64 * 256 + 64 * 256) * 2;                     // 64 KB
  const size_t blkLds = (size_t)(128*192 + 64*192 + 128*64 + 64*64) * 2
                      + (size_t)(128 * 64) * 4;                                // 128 KB
  const size_t upLds  = (size_t)(64*64 + N_ATOMS*CHUNK_T) * 4 + 256 * 8;       // ~146 KB
  const size_t tkLds  = (size_t)(TLEN / CHUNK_T) * KEEP * 8 + 1024 * 8;        // ~264 KB

  const int nPrep = NBLOCKS * 128 * 192 + NBLOCKS * 128 * 64 + 64 * 256;
  wprep_kernel<<<(nPrep + 255) / 256, blk, 0, stream>>>(sw, gw, ow, nw, fbw, wA1, wA2, wFb);

  fb_kernel<<<gT, blk, fbLds, stream>>>(x, wFb, actA);
  zero_kernel<<<(unsigned)((nTot + 255) / 256), blk, 0, stream>>>(outAcc, nTot);

  float* acts[2] = {actA, actB};
  int p = 0;
  for (int i = 0; i < 7; ++i) {                           // encode stack
    block_kernel<<<gT, blk, blkLds, stream>>>(acts[p], acts[p ^ 1], outAcc,
        wA1 + (size_t)i * 128 * 192, wA2 + (size_t)i * 128 * 64,
        sb + i * 64, gb + i * 64, ob + i * 64, nb + i * 64, DILS_[i]);
    p ^= 1;
  }

  upsel_kernel<<<BATCH * (TLEN / CHUNK_T), blk, upLds, stream>>>(outAcc, upw, upb, candV, candI);
  topk_final<<<BATCH, dim3(1024), tkLds, stream>>>(candV, candI, topV, topI);
  down_kernel<<<gT, blk, 0, stream>>>(topV, topI, dw, db, actA);

  zero_kernel<<<(unsigned)((nTot + 255) / 256), blk, 0, stream>>>(outAcc, nTot);
  p = 0;
  for (int i = 7; i < NBLOCKS; ++i) {                     // decode stack
    block_kernel<<<gT, blk, blkLds, stream>>>(acts[p], acts[p ^ 1], outAcc,
        wA1 + (size_t)i * 128 * 192, wA2 + (size_t)i * 128 * 64,
        sb + i * 64, gb + i * 64, ob + i * 64, nb + i * 64, DILS_[i]);
    p ^= 1;
  }

  final_kernel<<<(unsigned)((BATCH * TLEN + 255) / 256), blk, 0, stream>>>(
      outAcc, out, (size_t)BATCH * TLEN);
}
